// SinkhornScorer_2542620639847
// MI455X (gfx1250) — matrix operand order repaired and checked by reading, not compiled, not run on hardware
//
#include <hip/hip_runtime.h>
#include <hip/hip_bf16.h>
#include <math.h>

typedef __attribute__((ext_vector_type(16))) _Float16 v16h;
typedef __attribute__((ext_vector_type(8)))  _Float16 v8h;
typedef __attribute__((ext_vector_type(8)))  float    v8f;

#define N_LANG 64
#define N_EX   4
#define N_IM   256          // n * n_ex
#define M_IM   49
#define M_PAD  64           // xn rows padded to 64 per image (zero-filled)
#define L_TOK  32
#define HDIM   512
#define MP1    50
#define LP1    33
#define REGZ   0.1f
#define INV_REG 10.0f
#define SINK_ITERS 10
#define INV_TEMP 10.0f
#define NEGV  -1000000.0f
#define NRM_EPS 1e-12f

// ---------------------------------------------------------------------------
// L2-normalize x rows into a ROW-PADDED f16 buffer (256 x 64 x 512); pad rows
// 49..63 are zero-filled so the GEMM needs no per-lane predication.
// One wave32 per padded row, 8 rows per block.
// ---------------------------------------------------------------------------
__global__ __launch_bounds__(256)
void l2norm_x_pad_kernel(const float* __restrict__ src,
                         _Float16* __restrict__ dst) {
  const int wave = threadIdx.x >> 5;
  const int lane = threadIdx.x & 31;
  const int row  = blockIdx.x * 8 + wave;     // padded row index (0..16383)
  const int img  = row >> 6;                  // 0..255
  const int m    = row & 63;                  // 0..63
  _Float16* q = dst + (size_t)row * HDIM + lane * 16;
  if (m >= M_IM) {
    v8h z = {};
    *(v8h*)(q)     = z;
    *(v8h*)(q + 8) = z;
    return;
  }
  const float* p = src + ((size_t)img * M_IM + m) * HDIM + lane * 16;
  float vbuf[16];
  float s = 0.f;
#pragma unroll
  for (int t = 0; t < 16; ++t) { vbuf[t] = p[t]; s += vbuf[t] * vbuf[t]; }
#pragma unroll
  for (int off = 16; off > 0; off >>= 1) s += __shfl_xor(s, off, 32);
  const float inv = 1.0f / fmaxf(sqrtf(s), NRM_EPS);
#pragma unroll
  for (int t = 0; t < 16; ++t) q[t] = (_Float16)(vbuf[t] * inv);
}

// ---------------------------------------------------------------------------
// L2-normalize y rows (64 x 32 x 512), no padding needed.
// ---------------------------------------------------------------------------
__global__ __launch_bounds__(256)
void l2norm_y_kernel(const float* __restrict__ src,
                     _Float16* __restrict__ dst, int nrows) {
  const int wave = threadIdx.x >> 5;
  const int lane = threadIdx.x & 31;
  const int row  = blockIdx.x * 8 + wave;
  if (row >= nrows) return;
  const float* p = src + (size_t)row * HDIM + lane * 16;
  float vbuf[16];
  float s = 0.f;
#pragma unroll
  for (int t = 0; t < 16; ++t) { vbuf[t] = p[t]; s += vbuf[t] * vbuf[t]; }
#pragma unroll
  for (int off = 16; off > 0; off >>= 1) s += __shfl_xor(s, off, 32);
  const float inv = 1.0f / fmaxf(sqrtf(s), NRM_EPS);
  _Float16* q = dst + (size_t)row * HDIM + lane * 16;
#pragma unroll
  for (int t = 0; t < 16; ++t) q[t] = (_Float16)(vbuf[t] * inv);
}

// ---------------------------------------------------------------------------
// Fused per-(image,lang) pair kernel:
//   WMMA GEMM (64x32 padded, K=512, f16 in / f32 acc) -> LDS couplings ->
//   log-domain Sinkhorn (10 iters) -> transport plan to d_out + ot score.
// One block per (i,j); 256 threads = 8 waves; each wave owns a 16x16 C tile.
// All fragment loads AND the C->LDS stores are unconditional (row-padded).
// ---------------------------------------------------------------------------
__global__ __launch_bounds__(256)
void sinkhorn_fused_kernel(const _Float16* __restrict__ xn,   // (256,64,512)
                           const _Float16* __restrict__ yn,   // (64,32,512)
                           const float* __restrict__ dustbin_im,
                           const float* __restrict__ dustbin_lang,
                           const unsigned char* __restrict__ y_mask,
                           float* __restrict__ matching,
                           float* __restrict__ scores2) {
  __shared__ float S[M_PAD][LP1];      // rows 50..63 are scratch (never read)
  __shared__ float U[MP1], V[LP1], LMU[MP1], LNU[LP1];
  __shared__ int   MK[LP1];
  __shared__ float red[256];
  __shared__ float sh_ns;

  const int tid  = threadIdx.x;
  const int lane = tid & 31;
  const int wave = tid >> 5;
  const int b    = blockIdx.x;
  const int i    = b >> 6;     // image index (0..255)
  const int j    = b & 63;     // lang index  (0..63)

  // ---- Phase 1: scores via 8 WMMA tiles over the padded 64x32 block ----
  {
    const int mt  = wave >> 1;           // 0..3  M tile
    const int nt  = wave & 1;            // 0..1  N tile
    const int r16 = lane & 15;
    const int hi  = lane >> 4;           // 0 or 1
    const int arow = mt * 16 + r16;      // 0..63 (rows >=49 are zero data)
    const _Float16* Abase =
        xn + ((size_t)i * M_PAD + arow) * HDIM + hi * 16;
    const int bcol = nt * 16 + r16;      // 0..31
    const _Float16* Bbase =
        yn + ((size_t)j * L_TOK + bcol) * HDIM + hi * 16;

    v8f acc = {};
#pragma unroll 4
    for (int kk = 0; kk < HDIM; kk += 32) {
      // A frag: lane holds K = {kk+hi*16 .. +15}
      const v8h* pA = (const v8h*)(Abase + kk);
      v8h alo = pA[0];
      v8h ahi = pA[1];
      v16h A = __builtin_shufflevector(alo, ahi,
          0,1,2,3,4,5,6,7,8,9,10,11,12,13,14,15);
      // B frag: lane holds K = {kk+hi*16 .. +15} for column bcol
      const v8h* pB = (const v8h*)(Bbase + kk);
      v8h blo = pB[0];
      v8h bhi = pB[1];
      v16h B = __builtin_shufflevector(blo, bhi,
          0,1,2,3,4,5,6,7,8,9,10,11,12,13,14,15);
      acc = __builtin_amdgcn_wmma_f32_16x16x32_f16(
          /*neg_a=*/false, A, /*neg_b=*/false, B,
          /*c_mod=*/(short)0, acc, /*reuse_a=*/false, /*reuse_b=*/false);
    }
    // unconditional stores into the row-padded LDS tile
    const int crow0 = mt * 16 + hi * 8;
    const int ccol  = nt * 16 + r16;
#pragma unroll
    for (int vv = 0; vv < 8; ++vv)
      S[crow0 + vv][ccol] = acc[vv];
  }

  const float a_im   = fminf(fmaxf(dustbin_im[0],   -1.f), 1.f);
  const float a_lang = fminf(fmaxf(dustbin_lang[0], -1.f), 1.f);

  if (tid < LP1)
    MK[tid] = (tid < L_TOK) ? (y_mask[j * L_TOK + tid] ? 1 : 0) : 0;
  __syncthreads();

  // ---- Phase 2a: build masked couplings (append dustbin row/col) ----
  for (int idx = tid; idx < MP1 * LP1; idx += 256) {
    const int r = idx / LP1, c = idx % LP1;
    float val;
    if (r < M_IM) val = (c < L_TOK) ? S[r][c] : a_im;
    else          val = (c < L_TOK) ? a_lang  : a_im;
    if (c < L_TOK && MK[c]) val = NEGV;
    S[r][c] = val;
  }
  if (tid == 0) {
    int cnt = 0;
    for (int c = 0; c < L_TOK; ++c) cnt += MK[c] ? 0 : 1;
    sh_ns = (float)cnt;                 // valid tokens for this lang
  }
  __syncthreads();

  const float ns   = sh_ns;
  const float norm = -logf((float)M_IM + ns);

  if (tid < MP1) {
    LMU[tid] = (tid < M_IM) ? norm : (logf(ns) + norm);
    U[tid]   = 0.f;
  }
  if (tid < LP1) {
    const int msk = MK[tid];
    LNU[tid] = (tid < L_TOK) ? (msk ? NEGV : norm) : (logf((float)M_IM) + norm);
    V[tid]   = msk ? NEGV : 0.f;
  }
  __syncthreads();

  // ---- Phase 2b: log-domain Sinkhorn, 10 iterations ----
  for (int it = 0; it < SINK_ITERS; ++it) {
    if (tid < MP1) {                    // row logsumexp over 33 cols
      const float uu = U[tid];
      float mx = -INFINITY;
      for (int c = 0; c < LP1; ++c)
        mx = fmaxf(mx, (S[tid][c] + uu + V[c]) * INV_REG);
      float sm = 0.f;
      for (int c = 0; c < LP1; ++c)
        sm += expf((S[tid][c] + uu + V[c]) * INV_REG - mx);
      U[tid] = uu + REGZ * (LMU[tid] - (mx + logf(sm)));
    }
    __syncthreads();
    if (tid < LP1 && !MK[tid]) {        // col logsumexp over 50 rows
      const float vv = V[tid];
      float mx = -INFINITY;
      for (int r = 0; r < MP1; ++r)
        mx = fmaxf(mx, (S[r][tid] + U[r] + vv) * INV_REG);
      float sm = 0.f;
      for (int r = 0; r < MP1; ++r)
        sm += expf((S[r][tid] + U[r] + vv) * INV_REG - mx);
      V[tid] = vv + REGZ * (LNU[tid] - (mx + logf(sm)));
    }
    __syncthreads();
  }

  // ---- Phase 3: transport plan out + ot score reduction ----
  float ot = 0.f;
  float* outp = matching + (size_t)b * (MP1 * LP1);
  for (int idx = tid; idx < MP1 * LP1; idx += 256) {
    const int r = idx / LP1, c = idx % LP1;
    const float zm = expf((S[r][c] + U[r] + V[c]) * INV_REG - norm);
    outp[idx] = zm;                     // masked entries underflow to 0 exactly
    if (r < M_IM && c < L_TOK) ot += S[r][c] * zm;
  }
  red[tid] = ot;
  __syncthreads();
  for (int off = 128; off > 0; off >>= 1) {
    if (tid < off) red[tid] += red[tid + off];
    __syncthreads();
  }
  if (tid == 0) scores2[b] = red[0] * INV_TEMP;
}

// ---------------------------------------------------------------------------
// InfoNCE loss from scores2 (256 x 64). Single block, thread i = image i.
// ---------------------------------------------------------------------------
__global__ __launch_bounds__(256)
void loss_kernel(const float* __restrict__ sc2, float* __restrict__ out_loss) {
  __shared__ float rl[256], ri[256];
  const int i = threadIdx.x;
  const int g = i >> 2;                         // grp = i // n_ex
  const float pos = sc2[i * N_LANG + g];

  float mx = -INFINITY;
  for (int c = 0; c < N_LANG; ++c) mx = fmaxf(mx, sc2[i * N_LANG + c]);
  float sm = 0.f;
  for (int c = 0; c < N_LANG; ++c) sm += expf(sc2[i * N_LANG + c] - mx);
  const float lse_row = mx + logf(sm);

  float mx2 = -INFINITY;
  for (int k = 0; k < N_IM; ++k) {
    const bool allowed = ((k >> 2) != g) || (k == i);
    if (allowed) mx2 = fmaxf(mx2, sc2[k * N_LANG + g]);
  }
  float sm2 = 0.f;
  for (int k = 0; k < N_IM; ++k) {
    const bool allowed = ((k >> 2) != g) || (k == i);
    if (allowed) sm2 += expf(sc2[k * N_LANG + g] - mx2);
  }
  const float lse_im = mx2 + logf(sm2);

  rl[i] = lse_row - pos;   // -(pos - lse) terms
  ri[i] = lse_im  - pos;
  __syncthreads();
  for (int off = 128; off > 0; off >>= 1) {
    if (i < off) { rl[i] += rl[i + off]; ri[i] += ri[i + off]; }
    __syncthreads();
  }
  if (i == 0)
    *out_loss = 0.5f * (rl[0] / (float)N_IM) + 0.5f * (ri[0] / (float)N_IM);
}

// ---------------------------------------------------------------------------
extern "C" void kernel_launch(void* const* d_in, const int* in_sizes, int n_in,
                              void* d_out, int out_size, void* d_ws, size_t ws_size,
                              hipStream_t stream) {
  (void)in_sizes; (void)n_in; (void)ws_size;
  const float* x     = (const float*)d_in[0];           // (256, 49, 512)
  const float* y     = (const float*)d_in[1];           // (64, 32, 512)
  const float* d_im  = (const float*)d_in[2];           // scalar
  const float* d_lg  = (const float*)d_in[3];           // scalar
  const unsigned char* ymask = (const unsigned char*)d_in[4]; // (64, 32) bool
  float* out = (float*)d_out;   // [matching (256*64*50*33) | loss (1)]

  // workspace layout: xn_pad f16 (256*64*512) | yn f16 | scores2 f32 (~19 MB)
  _Float16* xn = (_Float16*)d_ws;
  const size_t NX = (size_t)N_IM * M_PAD * HDIM;
  _Float16* yn = xn + NX;
  const size_t NY = (size_t)N_LANG * L_TOK * HDIM;
  float* sc2 = (float*)(yn + NY);

  l2norm_x_pad_kernel<<<(N_IM * M_PAD) / 8, 256, 0, stream>>>(x, xn);
  l2norm_y_kernel<<<(N_LANG * L_TOK) / 8, 256, 0, stream>>>(y, yn, N_LANG * L_TOK);
  sinkhorn_fused_kernel<<<N_IM * N_LANG, 256, 0, stream>>>(
      xn, yn, d_im, d_lg, ymask, out, sc2);
  loss_kernel<<<1, 256, 0, stream>>>(sc2, out + (out_size - 1));
}
